// Encoder_21457656611103
// MI455X (gfx1250) — matrix operand order repaired and checked by reading, not compiled, not run on hardware
//
#include <hip/hip_runtime.h>

typedef unsigned short u16;
typedef __bf16 bf16;
typedef bf16 v16bf __attribute__((ext_vector_type(16)));
typedef float v8f __attribute__((ext_vector_type(8)));

union Frag32B { uint4 q[2]; v16bf v; };

__device__ __forceinline__ u16 f2bf(float f) {
  unsigned u = __float_as_uint(f);
  unsigned r = u + 0x7FFFu + ((u >> 16) & 1u);   // round-to-nearest-even
  return (u16)(r >> 16);
}

// ---------------------------------------------------------------------------
// WMMA bf16 GEMM: C[M,N] = A[M,K] @ Bt[N,K]^T (+bias, act).
// Workgroup: 4 waves, 32x128 tile, K step 64. Each wave: 16x64 strip =
// 4 accumulators; per 32-K substep one A fragment feeds 4 WMMAs.
// Fragment layouts per CDNA5 ISA 7.12.2.
// ---------------------------------------------------------------------------
__global__ __launch_bounds__(128) void gemm_bf16_kernel(
    const u16* __restrict__ A, const u16* __restrict__ Bt,
    const float* __restrict__ bias, float* __restrict__ Cf,
    u16* __restrict__ Cb, int M, int N, int K, int act) {
  __shared__ __align__(16) u16 lA[32 * 64];    // 4 KB
  __shared__ __align__(16) u16 lB[128 * 64];   // 16 KB
  const int tid = threadIdx.x;
  const int wave = tid >> 5;
  const int lane = tid & 31;
  const int mBase = blockIdx.y * 32;
  const int nBase = blockIdx.x * 128;
  const int wm = (wave & 1) << 4;   // wave row offset within 32
  const int wn = (wave >> 1) << 6;  // wave col offset within 128 (0 / 64)
  v8f acc[4];
#pragma unroll
  for (int j = 0; j < 4; j++) acc[j] = v8f{0.f, 0.f, 0.f, 0.f, 0.f, 0.f, 0.f, 0.f};

  const int fr = wm + (lane & 15);       // A frag row
  const int k8 = (lane & 16) ? 8 : 0;    // A frag K base
  const int bc = wn + (lane & 15);       // B frag col base
  const int klo = (lane & 16) ? 8 : 0;   // B frag K base

  for (int k0 = 0; k0 < K; k0 += 64) {
    // stage A tile (32x64 bf16): 256 16B chunks, coalesced, zero-pad rows >= M
#pragma unroll
    for (int i = 0; i < 2; i++) {
      int c = i * 128 + tid;
      int row = c >> 3, seg = (c & 7) << 3;  // u16 offset
      uint4 v = {0u, 0u, 0u, 0u};
      if (mBase + row < M)
        v = *(const uint4*)(A + (size_t)(mBase + row) * K + k0 + seg);
      *(uint4*)(lA + row * 64 + seg) = v;
    }
    // stage B tile (128x64 bf16) from pre-transposed weights: 1024 chunks
#pragma unroll
    for (int i = 0; i < 8; i++) {
      int c = i * 128 + tid;
      int row = c >> 3, seg = (c & 7) << 3;
      uint4 v = *(const uint4*)(Bt + (size_t)(nBase + row) * K + k0 + seg);
      *(uint4*)(lB + row * 64 + seg) = v;
    }
    if (k0 + 64 < K) {  // global_prefetch_b8 hints for next K tile
      int row = tid >> 3, seg = (tid & 7) << 3;
      if (mBase + row < M)
        __builtin_prefetch(A + (size_t)(mBase + row) * K + k0 + 64 + seg, 0, 1);
      __builtin_prefetch(Bt + (size_t)(nBase + tid) * K + k0 + 64, 0, 1);
    }
    __syncthreads();
#pragma unroll
    for (int kk = 0; kk < 64; kk += 32) {
      Frag32B fa;
      // A: elems 0..7 -> K=kk+k8.., elems 8..15 -> K=kk+k8+16..
      fa.q[0] = *(const uint4*)(lA + fr * 64 + kk + k8);
      fa.q[1] = *(const uint4*)(lA + fr * 64 + kk + k8 + 16);
#pragma unroll
      for (int j = 0; j < 4; j++) {
        Frag32B fb;  // B: lane=col, elems 0..7 -> K=kk+klo.., elems 8..15 -> K=kk+klo+16..
        const u16* p = lB + (bc + j * 16) * 64 + kk + klo;
        fb.q[0] = *(const uint4*)(p);
        fb.q[1] = *(const uint4*)(p + 16);
        acc[j] = __builtin_amdgcn_wmma_f32_16x16x32_bf16(
            false, fa.v, false, fb.v, (short)0, acc[j], false, false);
      }
    }
    __syncthreads();
  }
  const int rb = mBase + wm + ((lane & 16) ? 8 : 0);  // C/D: M = e + 8*(lane>=16)
#pragma unroll
  for (int j = 0; j < 4; j++) {
    int nc = nBase + wn + j * 16 + (lane & 15);
    float bi = bias ? bias[nc] : 0.f;
#pragma unroll
    for (int e = 0; e < 8; e++) {
      int m = rb + e;
      if (m < M) {
        float v = acc[j][e] + bi;
        if (act == 1) v = fmaxf(v, 0.f);
        size_t o = (size_t)m * N + nc;
        Cf[o] = v;
        if (Cb) Cb[o] = f2bf(v);
      }
    }
  }
}

// W[K,N] f32 -> Bt[N,K] bf16, 16x16 LDS tile transpose (K,N multiples of 16)
__global__ __launch_bounds__(256) void wt_transpose_kernel(
    const float* __restrict__ W, u16* __restrict__ Bt, int K, int N) {
  __shared__ u16 t[16][17];
  int n0 = blockIdx.x << 4, k0 = blockIdx.y << 4;
  int tx = threadIdx.x & 15, ty = threadIdx.x >> 4;
  t[ty][tx] = f2bf(W[(size_t)(k0 + ty) * N + n0 + tx]);
  __syncthreads();
  Bt[(size_t)(n0 + ty) * K + k0 + tx] = t[tx][ty];
}

__global__ __launch_bounds__(256) void f32_to_bf16_kernel(
    const float* __restrict__ s, u16* __restrict__ d, int n) {
  int i = blockIdx.x * 256 + threadIdx.x;
  if (i < n) d[i] = f2bf(s[i]);
}

// ---------------------------------------------------------------------------
// Conv 3x3 SAME + leaky + maxpool 2x2 + BN(inference) fused. pfold=1 folds the
// predictor axis (conv1: channel stride 3, offset p = nt/128).
// ---------------------------------------------------------------------------
template <int CI, int CO>
__global__ __launch_bounds__(256) void convpool_kernel(
    const float* __restrict__ xin, const float* __restrict__ W,
    const float* __restrict__ bias, float* __restrict__ y, int NT, int H,
    int Wd, int cstride, int pfold) {
  const int Ho = H >> 1, Wo = Wd >> 1;
  size_t id = (size_t)blockIdx.x * 256 + threadIdx.x;
  size_t total = (size_t)NT * Ho * Wo;
  if (id >= total) return;
  int wo = (int)(id % Wo);
  size_t r = id / Wo;
  int ho = (int)(r % Ho);
  int nt = (int)(r / Ho);
  int in_nt = nt, coff = 0;
  if (pfold) { coff = nt >> 7; in_nt = nt & 127; }
  const int pstr = CI * cstride;
  const size_t plane = (size_t)in_nt * H * Wd;
  float mx[CO];
#pragma unroll
  for (int c = 0; c < CO; c++) mx[c] = -3.4e38f;
#pragma unroll
  for (int i = 0; i < 2; i++) {
#pragma unroll
    for (int j = 0; j < 2; j++) {
      float acc[CO];
#pragma unroll
      for (int c = 0; c < CO; c++) acc[c] = bias[c];
      int hc = 2 * ho + i - 1, wc = 2 * wo + j - 1;
      for (int dy = 0; dy < 3; dy++) {
        int h = hc + dy;
        if (h < 0 || h >= H) continue;
        for (int dx = 0; dx < 3; dx++) {
          int w = wc + dx;
          if (w < 0 || w >= Wd) continue;
          const float* px = xin + (plane + (size_t)h * Wd + w) * pstr + coff;
          const float* wp = W + ((dy * 3 + dx) * CI) * CO;
#pragma unroll
          for (int ci = 0; ci < CI; ci++) {
            float v = px[ci * cstride];
#pragma unroll
            for (int c = 0; c < CO; c++) acc[c] = fmaf(v, wp[ci * CO + c], acc[c]);
          }
        }
      }
#pragma unroll
      for (int c = 0; c < CO; c++) {
        float z = acc[c];
        z = (z >= 0.f) ? z : 0.3f * z;
        mx[c] = fmaxf(mx[c], z);
      }
    }
  }
  const float bnscale = 0.999500373f;  // 1/sqrt(1+1e-3)
  float* yo = y + ((size_t)nt * Ho * Wo + (size_t)ho * Wo + wo) * CO;
#pragma unroll
  for (int c = 0; c < CO; c++) yo[c] = mx[c] * bnscale;
}

// 5x5 stride-5 VALID conv 16->32 + leaky: [384,20,40,16] -> [384,4,8,32]
__global__ __launch_bounds__(256) void conv_final_kernel(
    const float* __restrict__ yin, const float* __restrict__ W,
    const float* __restrict__ bias, float* __restrict__ z) {
  int id = blockIdx.x * 256 + threadIdx.x;
  if (id >= 384 * 4 * 8 * 32) return;
  int co = id & 31;
  int r = id >> 5;
  int ox = r & 7; r >>= 3;
  int oy = r & 3;
  int nt = r >> 2;
  float acc = bias[co];
  for (int ky = 0; ky < 5; ky++)
    for (int kx = 0; kx < 5; kx++) {
      const float* ip = yin + (((size_t)nt * 20 + oy * 5 + ky) * 40 + ox * 5 + kx) * 16;
      const float* wp = W + ((ky * 5 + kx) * 16) * 32 + co;
#pragma unroll
      for (int ci = 0; ci < 16; ci++) acc = fmaf(ip[ci], wp[ci * 32], acc);
    }
  z[id] = (acc >= 0.f) ? acc : 0.3f * acc;
}

// score[nt] = tanh(flat . W1 + b1) . W2 + b2 ; one workgroup per (n,t)
__global__ __launch_bounds__(256) void attn_score_kernel(
    const float* __restrict__ flat, const float* __restrict__ W1,
    const float* __restrict__ b1, const float* __restrict__ W2,
    const float* __restrict__ b2, float* __restrict__ score, int D) {
  __shared__ float red[16 * 256];
  const int tid = threadIdx.x;
  const float* f = flat + (size_t)blockIdx.x * D;
  float acc[16];
#pragma unroll
  for (int h = 0; h < 16; h++) acc[h] = 0.f;
  for (int d = tid; d < D; d += 256) {
    float v = f[d];
    const float* w = W1 + (size_t)d * 16;
#pragma unroll
    for (int h = 0; h < 16; h++) acc[h] = fmaf(v, w[h], acc[h]);
  }
#pragma unroll
  for (int h = 0; h < 16; h++) red[h * 256 + tid] = acc[h];
  __syncthreads();
  for (int s = 128; s > 0; s >>= 1) {
    if (tid < s) {
#pragma unroll
      for (int h = 0; h < 16; h++) red[h * 256 + tid] += red[h * 256 + tid + s];
    }
    __syncthreads();
  }
  if (tid == 0) {
    float sc = b2[0];
#pragma unroll
    for (int h = 0; h < 16; h++) sc += tanhf(red[h * 256] + b1[h]) * W2[h];
    score[blockIdx.x] = sc;
  }
}

// skip[n,d] = sum_t softmax_t(score[n,:]) * flat[n,t,d]
__global__ __launch_bounds__(256) void attn_wsum_kernel(
    const float* __restrict__ flat, const float* __restrict__ score,
    float* __restrict__ skip, int D) {
  size_t id = (size_t)blockIdx.x * 256 + threadIdx.x;
  size_t total = (size_t)24 * D;
  if (id >= total) return;
  int d = (int)(id % D);
  int n = (int)(id / D);
  const float* s = score + n * 16;
  float m = -3.4e38f;
#pragma unroll
  for (int t = 0; t < 16; t++) m = fmaxf(m, s[t]);
  float e[16];
  float sum = 0.f;
#pragma unroll
  for (int t = 0; t < 16; t++) { e[t] = __expf(s[t] - m); sum += e[t]; }
  float inv = 1.f / sum;
  float acc = 0.f;
#pragma unroll
  for (int t = 0; t < 16; t++) acc += e[t] * flat[((size_t)n * 16 + t) * D + d];
  skip[id] = acc * inv;
}

// xcur[b,t,m,f] = z[(m*8+b)*16+t, f] + posenc(t,f)
__global__ __launch_bounds__(256) void emb_kernel(const float* __restrict__ z,
                                                  float* __restrict__ xc) {
  int id = blockIdx.x * 256 + threadIdx.x;
  if (id >= 8 * 16 * 3 * 1024) return;
  int f = id & 1023;
  int r = id >> 10;
  int m = r % 3; r /= 3;
  int t = r & 15;
  int b = r >> 4;
  float i2 = (float)((f >> 1) << 1);
  float ang = (float)t * __expf(-(i2 / 1024.f) * 9.21034037f);  // ln(1e4)
  float pe = (f & 1) ? __cosf(ang) : __sinf(ang);
  xc[id] = z[(((size_t)(m * 8 + b) * 16 + t) << 10) + f] + pe;
}

// xm_bf[(b*3+m), t*1024+f] = bf16(xcur[b,t,m,f])
__global__ __launch_bounds__(256) void xm_convert_kernel(
    const float* __restrict__ xc, u16* __restrict__ xm) {
  int id = blockIdx.x * 256 + threadIdx.x;
  if (id >= 8 * 3 * 16 * 1024) return;
  int f = id & 1023;
  int r = id >> 10;
  int t = r & 15; r >>= 4;
  int m = r % 3;
  int b = r / 3;
  xm[id] = f2bf(xc[(((size_t)(b * 16 + t) * 3 + m) << 10) + f]);
}

// im2col for 3x3 SAME conv over the (T=16, M=3) grid: col[384, 9216] bf16
__global__ __launch_bounds__(256) void im2col_kernel(
    const float* __restrict__ xc, u16* __restrict__ col) {
  size_t id = (size_t)blockIdx.x * 256 + threadIdx.x;
  if (id >= (size_t)384 * 9216) return;
  int c = (int)(id % 9216);
  int r = (int)(id / 9216);
  int m = r % 3;
  int t = (r / 3) & 15;
  int b = r / 48;
  int f = c & 1023;
  int tap = c >> 10;
  int dy = tap / 3 - 1, dx = tap % 3 - 1;
  int tt = t + dy, mm = m + dx;
  float v = 0.f;
  if (tt >= 0 && tt < 16 && mm >= 0 && mm < 3)
    v = xc[(((size_t)(b * 16 + tt) * 3 + mm) << 10) + f];
  col[id] = f2bf(v);
}

// At[b,h,q,k] = softmax_k(Qt.Kt/8), Tu=64
__global__ __launch_bounds__(256) void at_kernel(const float* __restrict__ Qt,
                                                 const float* __restrict__ Kt,
                                                 float* __restrict__ At) {
  int id = blockIdx.x * 256 + threadIdx.x;
  if (id >= 1024) return;
  int q = id & 15;
  int h = (id >> 4) & 7;
  int b = id >> 7;
  const float* qp = Qt + (size_t)(b * 16 + q) * 512 + h * 64;
  float lg[16];
  float mx = -3.4e38f;
  for (int k = 0; k < 16; k++) {
    const float* kp = Kt + (size_t)(b * 16 + k) * 512 + h * 64;
    float s = 0.f;
    for (int d = 0; d < 64; d++) s = fmaf(qp[d], kp[d], s);
    lg[k] = s * 0.125f;
    mx = fmaxf(mx, lg[k]);
  }
  float sum = 0.f;
  for (int k = 0; k < 16; k++) { lg[k] = __expf(lg[k] - mx); sum += lg[k]; }
  float inv = 1.f / sum;
  float* op = At + (size_t)((b * 8 + h) * 16 + q) * 16;
  for (int k = 0; k < 16; k++) op[k] = lg[k] * inv;
}

// Am[b,h,n,m] = softmax_m(Qm.Km/8), Mu=64
__global__ __launch_bounds__(256) void am_kernel(const float* __restrict__ Qm,
                                                 const float* __restrict__ Km,
                                                 float* __restrict__ Am) {
  int id = blockIdx.x * 256 + threadIdx.x;
  if (id >= 192) return;
  int n = id % 3;
  int h = (id / 3) & 7;
  int b = id / 24;
  const float* qp = Qm + (size_t)(b * 3 + n) * 512 + h * 64;
  float lg[3];
  float mx = -3.4e38f;
  for (int m = 0; m < 3; m++) {
    const float* kp = Km + (size_t)(b * 3 + m) * 512 + h * 64;
    float s = 0.f;
    for (int d = 0; d < 64; d++) s = fmaf(qp[d], kp[d], s);
    lg[m] = s * 0.125f;
    mx = fmaxf(mx, lg[m]);
  }
  float sum = 0.f;
  for (int m = 0; m < 3; m++) { lg[m] = __expf(lg[m] - mx); sum += lg[m]; }
  float inv = 1.f / sum;
  float* op = Am + (size_t)((b * 8 + h) * 3 + n) * 3;
  for (int m = 0; m < 3; m++) op[m] = lg[m] * inv;
}

// O[b,q,n,h,v] = sum_m Am * sum_k At * V[b,k,m,h,v]  (dv = 128)
__global__ __launch_bounds__(256) void o_kernel(const float* __restrict__ At,
                                                const float* __restrict__ Am,
                                                const float* __restrict__ V,
                                                float* __restrict__ O) {
  int id = blockIdx.x * 256 + threadIdx.x;
  if (id >= 393216) return;
  int v = id & 127;
  int h = (id >> 7) & 7;
  int r = id >> 10;
  int n = r % 3;
  int r2 = r / 3;
  int q = r2 & 15;
  int b = r2 >> 4;
  const float* atp = At + (size_t)((b * 8 + h) * 16 + q) * 16;
  const float* amp = Am + (size_t)((b * 8 + h) * 3 + n) * 3;
  float acc = 0.f;
  for (int m = 0; m < 3; m++) {
    float t = 0.f;
    for (int k = 0; k < 16; k++)
      t = fmaf(atp[k], V[((size_t)(b * 48 + k * 3 + m) << 10) + h * 128 + v], t);
    acc = fmaf(amp[m], t, acc);
  }
  O[id] = acc;
}

// out = LN(a + bsrc)*g + beta ; optional bf16 mirror. One block per row (1024)
__global__ __launch_bounds__(256) void ln_kernel(
    const float* __restrict__ a, const float* __restrict__ bsrc,
    const float* __restrict__ g, const float* __restrict__ beta,
    float* __restrict__ outf, u16* __restrict__ outb) {
  __shared__ float rs[256], rq[256];
  int row = blockIdx.x;
  int tid = threadIdx.x;
  const float* ap = a + (size_t)row * 1024;
  const float* bp = bsrc + (size_t)row * 1024;
  float loc[4];
  float s = 0.f, s2 = 0.f;
#pragma unroll
  for (int i = 0; i < 4; i++) {
    float v = ap[tid + i * 256] + bp[tid + i * 256];
    loc[i] = v; s += v; s2 += v * v;
  }
  rs[tid] = s; rq[tid] = s2;
  __syncthreads();
  for (int st = 128; st > 0; st >>= 1) {
    if (tid < st) { rs[tid] += rs[tid + st]; rq[tid] += rq[tid + st]; }
    __syncthreads();
  }
  float mean = rs[0] * (1.f / 1024.f);
  float var = rq[0] * (1.f / 1024.f) - mean * mean;
  float inv = rsqrtf(var + 1e-6f);
#pragma unroll
  for (int i = 0; i < 4; i++) {
    int c = tid + i * 256;
    float yv = (loc[i] - mean) * inv * g[c] + beta[c];
    outf[(size_t)row * 1024 + c] = yv;
    if (outb) outb[(size_t)row * 1024 + c] = f2bf(yv);
  }
}

static inline void gemm(const u16* A, const u16* Bt, const float* bias,
                        float* Cf, u16* Cb, int M, int N, int K, int act,
                        hipStream_t s) {
  dim3 g(N / 128, (M + 31) / 32);
  gemm_bf16_kernel<<<g, 128, 0, s>>>(A, Bt, bias, Cf, Cb, M, N, K, act);
}

extern "C" void kernel_launch(void* const* d_in, const int* in_sizes, int n_in,
                              void* d_out, int out_size, void* d_ws,
                              size_t ws_size, hipStream_t stream) {
  (void)in_sizes; (void)n_in; (void)out_size; (void)ws_size;
  const float* x    = (const float*)d_in[0];
  const float* c1W  = (const float*)d_in[1];
  const float* c1b  = (const float*)d_in[2];
  const float* a1W1 = (const float*)d_in[3];
  const float* a1b1 = (const float*)d_in[4];
  const float* a1W2 = (const float*)d_in[5];
  const float* a1b2 = (const float*)d_in[6];
  const float* c2W  = (const float*)d_in[7];
  const float* c2b  = (const float*)d_in[8];
  const float* a2W1 = (const float*)d_in[9];
  const float* a2b1 = (const float*)d_in[10];
  const float* a2W2 = (const float*)d_in[11];
  const float* a2b2 = (const float*)d_in[12];
  const float* c3W  = (const float*)d_in[13];
  const float* c3b  = (const float*)d_in[14];
  const float* a3W1 = (const float*)d_in[15];
  const float* a3b1 = (const float*)d_in[16];
  const float* a3W2 = (const float*)d_in[17];
  const float* a3b2 = (const float*)d_in[18];
  const float* fW   = (const float*)d_in[19];
  const float* fb   = (const float*)d_in[20];
  const float* qtW  = (const float*)d_in[21];
  const float* qtb  = (const float*)d_in[22];
  const float* ktW  = (const float*)d_in[23];
  const float* ktb  = (const float*)d_in[24];
  const float* qmW  = (const float*)d_in[25];
  const float* qmb  = (const float*)d_in[26];
  const float* kmW  = (const float*)d_in[27];
  const float* kmb  = (const float*)d_in[28];
  const float* vW   = (const float*)d_in[29];
  const float* vb   = (const float*)d_in[30];
  const float* oW   = (const float*)d_in[31];
  const float* ob   = (const float*)d_in[32];
  const float* fW1  = (const float*)d_in[33];
  const float* fb1  = (const float*)d_in[34];
  const float* fW2  = (const float*)d_in[35];
  const float* fb2  = (const float*)d_in[36];
  const float* ln1g = (const float*)d_in[37];
  const float* ln1b = (const float*)d_in[38];
  const float* ln2g = (const float*)d_in[39];
  const float* ln2b = (const float*)d_in[40];
  float* out = (float*)d_out;

  char* ws = (char*)d_ws;
  size_t off = 0;
  auto alloc = [&](size_t bytes) -> void* {
    void* p = ws + off;
    off = (off + bytes + 255) & ~(size_t)255;
    return p;
  };
  float* y1  = (float*)alloc(19660800ull * 4);   // [24,16,80,160,4]
  float* y2  = (float*)alloc(9830400ull * 4);    // [24,16,40,80,8]
  float* y3  = (float*)alloc(4915200ull * 4);    // [24,16,20,40,16]
  float* zb  = (float*)alloc(393216ull * 4);     // [384,1024]
  float* xc  = (float*)alloc(393216ull * 4);     // [8,16,3,1024]
  float* sc1 = (float*)alloc(384 * 4);
  float* sc2 = (float*)alloc(384 * 4);
  float* sc3 = (float*)alloc(384 * 4);
  u16* qtWt = (u16*)alloc(512ull * 3072 * 2);
  u16* ktWt = (u16*)alloc(512ull * 3072 * 2);
  u16* qmWt = (u16*)alloc(512ull * 16384 * 2);
  u16* kmWt = (u16*)alloc(512ull * 16384 * 2);
  u16* vWt  = (u16*)alloc(1024ull * 9216 * 2);
  u16* oWt  = (u16*)alloc(1024ull * 1024 * 2);
  u16* f1Wt = (u16*)alloc(2048ull * 1024 * 2);
  u16* f2Wt = (u16*)alloc(1024ull * 2048 * 2);
  u16* xt_b  = (u16*)alloc(128ull * 3072 * 2);
  u16* xm_b  = (u16*)alloc(24ull * 16384 * 2);
  u16* col_b = (u16*)alloc(384ull * 9216 * 2);
  float* Qt = (float*)alloc(128ull * 512 * 4);
  float* Kt = (float*)alloc(128ull * 512 * 4);
  float* Qm = (float*)alloc(24ull * 512 * 4);
  float* Km = (float*)alloc(24ull * 512 * 4);
  float* Vf = (float*)alloc(384ull * 1024 * 4);
  float* Atb = (float*)alloc(8ull * 8 * 16 * 16 * 4);
  float* Amb = (float*)alloc(8ull * 8 * 3 * 3 * 4);
  float* Of = (float*)alloc(384ull * 1024 * 4);
  u16* O_b  = (u16*)alloc(384ull * 1024 * 2);
  float* attnf = (float*)alloc(384ull * 1024 * 4);
  float* x1f   = (float*)alloc(384ull * 1024 * 4);
  u16* x1b     = (u16*)alloc(384ull * 1024 * 2);
  float* h1f   = (float*)alloc(384ull * 2048 * 4);
  u16* h1b     = (u16*)alloc(384ull * 2048 * 2);
  float* ffnf  = (float*)alloc(384ull * 1024 * 4);

  float* s1_out = out + 393216;
  float* s2_out = s1_out + 1228800;
  float* s3_out = s2_out + 614400;

  // ---- conv/pool/attention pyramid ----
  convpool_kernel<1, 4><<<(4915200 + 255) / 256, 256, 0, stream>>>(
      x, c1W, c1b, y1, 384, 160, 320, 3, 1);
  attn_score_kernel<<<384, 256, 0, stream>>>(y1, a1W1, a1b1, a1W2, a1b2, sc1, 51200);
  attn_wsum_kernel<<<(24 * 51200 + 255) / 256, 256, 0, stream>>>(y1, sc1, s1_out, 51200);
  convpool_kernel<4, 8><<<(1228800 + 255) / 256, 256, 0, stream>>>(
      y1, c2W, c2b, y2, 384, 80, 160, 1, 0);
  attn_score_kernel<<<384, 256, 0, stream>>>(y2, a2W1, a2b1, a2W2, a2b2, sc2, 25600);
  attn_wsum_kernel<<<(24 * 25600 + 255) / 256, 256, 0, stream>>>(y2, sc2, s2_out, 25600);
  convpool_kernel<8, 16><<<(307200 + 255) / 256, 256, 0, stream>>>(
      y2, c3W, c3b, y3, 384, 40, 80, 1, 0);
  attn_score_kernel<<<384, 256, 0, stream>>>(y3, a3W1, a3b1, a3W2, a3b2, sc3, 12800);
  attn_wsum_kernel<<<(24 * 12800 + 255) / 256, 256, 0, stream>>>(y3, sc3, s3_out, 12800);
  conv_final_kernel<<<(393216 + 255) / 256, 256, 0, stream>>>(y3, fW, fb, zb);
  emb_kernel<<<1536, 256, 0, stream>>>(zb, xc);

  // ---- transformer layers (WMMA GEMMs) ----
  for (int L = 0; L < 2; L++) {
    const float* qtWl = qtW + (size_t)L * 3072 * 512;
    const float* ktWl = ktW + (size_t)L * 3072 * 512;
    const float* qmWl = qmW + (size_t)L * 16384 * 512;
    const float* kmWl = kmW + (size_t)L * 16384 * 512;
    const float* vWl  = vW + (size_t)L * 9216 * 1024;
    const float* oWl  = oW + (size_t)L * 1024 * 1024;
    const float* fW1l = fW1 + (size_t)L * 1024 * 2048;
    const float* fW2l = fW2 + (size_t)L * 2048 * 1024;
    const float* qtbL = qtb + L * 512;
    const float* ktbL = ktb + L * 512;
    const float* qmbL = qmb + L * 512;
    const float* kmbL = kmb + L * 512;
    const float* vbL  = vb + L * 1024;
    const float* obL  = ob + L * 1024;
    const float* fb1L = fb1 + L * 2048;
    const float* fb2L = fb2 + L * 1024;
    const float* l1g = ln1g + L * 1024, *l1b = ln1b + L * 1024;
    const float* l2g = ln2g + L * 1024, *l2b = ln2b + L * 1024;

    wt_transpose_kernel<<<dim3(512 / 16, 3072 / 16), 256, 0, stream>>>(qtWl, qtWt, 3072, 512);
    wt_transpose_kernel<<<dim3(512 / 16, 3072 / 16), 256, 0, stream>>>(ktWl, ktWt, 3072, 512);
    wt_transpose_kernel<<<dim3(512 / 16, 16384 / 16), 256, 0, stream>>>(qmWl, qmWt, 16384, 512);
    wt_transpose_kernel<<<dim3(512 / 16, 16384 / 16), 256, 0, stream>>>(kmWl, kmWt, 16384, 512);
    wt_transpose_kernel<<<dim3(1024 / 16, 9216 / 16), 256, 0, stream>>>(vWl, vWt, 9216, 1024);
    wt_transpose_kernel<<<dim3(1024 / 16, 1024 / 16), 256, 0, stream>>>(oWl, oWt, 1024, 1024);
    wt_transpose_kernel<<<dim3(2048 / 16, 1024 / 16), 256, 0, stream>>>(fW1l, f1Wt, 1024, 2048);
    wt_transpose_kernel<<<dim3(1024 / 16, 2048 / 16), 256, 0, stream>>>(fW2l, f2Wt, 2048, 1024);

    f32_to_bf16_kernel<<<1536, 256, 0, stream>>>(xc, xt_b, 393216);
    gemm(xt_b, qtWt, qtbL, Qt, nullptr, 128, 512, 3072, 0, stream);
    gemm(xt_b, ktWt, ktbL, Kt, nullptr, 128, 512, 3072, 0, stream);
    xm_convert_kernel<<<1536, 256, 0, stream>>>(xc, xm_b);
    gemm(xm_b, qmWt, qmbL, Qm, nullptr, 24, 512, 16384, 0, stream);
    gemm(xm_b, kmWt, kmbL, Km, nullptr, 24, 512, 16384, 0, stream);
    im2col_kernel<<<(int)((384ull * 9216 + 255) / 256), 256, 0, stream>>>(xc, col_b);
    gemm(col_b, vWt, vbL, Vf, nullptr, 384, 1024, 9216, 0, stream);
    at_kernel<<<4, 256, 0, stream>>>(Qt, Kt, Atb);
    am_kernel<<<1, 256, 0, stream>>>(Qm, Km, Amb);
    o_kernel<<<1536, 256, 0, stream>>>(Atb, Amb, Vf, Of);
    f32_to_bf16_kernel<<<1536, 256, 0, stream>>>(Of, O_b, 393216);
    gemm(O_b, oWt, obL, attnf, nullptr, 384, 1024, 1024, 0, stream);
    ln_kernel<<<384, 256, 0, stream>>>(xc, attnf, l1g, l1b, x1f, x1b);
    gemm(x1b, f1Wt, fb1L, h1f, h1b, 384, 2048, 1024, 1, stream);
    gemm(h1b, f2Wt, fb2L, ffnf, nullptr, 384, 1024, 2048, 0, stream);
    ln_kernel<<<384, 256, 0, stream>>>(x1f, ffnf, l2g, l2b,
                                       (L == 1) ? out : xc, nullptr);
  }
}
